// EMTransformerLayer_35141422416176
// MI455X (gfx1250) — compile-verified, matrix operand order checked by reading, not run on hardware
//
#include <hip/hip_runtime.h>
#include <math.h>

// ---------------------------------------------------------------------------
// EMTransformer layer for MI455X (gfx1250, wave32, WMMA).
// bf16 operands, f32 accumulation via v_wmma_f32_16x16x32_bf16.
// A-tiles staged with global_load_async_to_lds_b128 (ASYNCcnt) + double buffer.
// ---------------------------------------------------------------------------

typedef __attribute__((ext_vector_type(16))) __bf16 v16bf;
typedef __attribute__((ext_vector_type(8)))  float  v8f;

static constexpr int kB  = 2;
static constexpr int kT  = 2048;
static constexpr int kD  = 1024;
static constexpr int kH  = 16;
static constexpr int kDH = 64;
static constexpr int kF  = 4096;   // 4*D
static constexpr int kBT = kB * kT;

__device__ inline v8f zero8() {
  v8f z;
#pragma unroll
  for (int i = 0; i < 8; ++i) z[i] = 0.f;
  return z;
}

__device__ inline v8f wmma_bf16(v16bf a, v16bf b, v8f c) {
  // D = A(16x32) * B(32x16) + C(16x16 f32)
  return __builtin_amdgcn_wmma_f32_16x16x32_bf16(
      /*neg_a=*/false, a, /*neg_b=*/false, b,
      /*c_mod=*/(short)0, c, /*reuse_a=*/false, /*reuse_b=*/false);
}

// --- CDNA5 async copy: global -> LDS, tracked by ASYNCcnt ------------------
// LDS byte offset of a __shared__ object: the addrspace(3) pointer value.
__device__ inline unsigned lds_addr32(const void* p) {
  return (unsigned)(unsigned long long)(__attribute__((address_space(3))) const char*)p;
}

__device__ inline void async_ld_64B(unsigned lds, const void* g) {
  const unsigned long long ga = (unsigned long long)(size_t)g;
  // INST_OFFSET is added to both the LDS and global addresses (ISA ch.10).
  asm volatile(
      "global_load_async_to_lds_b128 %0, %1, off\n\t"
      "global_load_async_to_lds_b128 %0, %1, off offset:16\n\t"
      "global_load_async_to_lds_b128 %0, %1, off offset:32\n\t"
      "global_load_async_to_lds_b128 %0, %1, off offset:48"
      :: "v"(lds), "v"(ga) : "memory");
}

__device__ inline void async_ld_32B(unsigned lds, const void* g) {
  const unsigned long long ga = (unsigned long long)(size_t)g;
  asm volatile(
      "global_load_async_to_lds_b128 %0, %1, off\n\t"
      "global_load_async_to_lds_b128 %0, %1, off offset:16"
      :: "v"(lds), "v"(ga) : "memory");
}

__device__ inline void wait_async0() {
  asm volatile("s_wait_asynccnt 0x0" ::: "memory");
}

// Load one 16x32 bf16 operand fragment from an LDS tile (ISA 7.12.2 layout).
// A: p -> tile[m0][k0]; lane L holds row m=L%16; VGPR r<4 packs K=2r,2r+1
// (+8 upper half-wave), VGPR r>=4 packs K=16+2(r-4) (+8 upper half-wave).
// B: same pattern with LDS stored as [n][k].
__device__ inline v16bf load_frag16(const __bf16* p, int ld) {
  const int lane = threadIdx.x & 31;
  const __bf16* r = p + (lane & 15) * ld + 8 * (lane >> 4);
  v16bf f;
#pragma unroll
  for (int q = 0; q < 4; ++q) {          // VGPRs 0..3
    f[2 * q]     = r[2 * q];
    f[2 * q + 1] = r[2 * q + 1];
  }
#pragma unroll
  for (int q = 0; q < 4; ++q) {          // VGPRs 4..7
    f[8 + 2 * q] = r[16 + 2 * q];
    f[9 + 2 * q] = r[16 + 2 * q + 1];
  }
  return f;
}

// ---------------------------------------------------------------------------
// fp32 -> bf16 cast (weights / pos_emb)
// ---------------------------------------------------------------------------
__global__ void cast_f32_bf16_kernel(const float* __restrict__ in,
                                     __bf16* __restrict__ out, int n) {
  for (int i = blockIdx.x * blockDim.x + threadIdx.x; i < n;
       i += gridDim.x * blockDim.x)
    out[i] = (__bf16)in[i];
}

// ---------------------------------------------------------------------------
// LayerNorm (one row of D=1024 per block) -> bf16
// ---------------------------------------------------------------------------
__global__ __launch_bounds__(256) void ln_bf16_kernel(
    const float* __restrict__ x, const float* __restrict__ w,
    const float* __restrict__ bvec, __bf16* __restrict__ out) {
  const int row = blockIdx.x;
  const float* xr = x + (size_t)row * kD;
  float s = 0.f, ss = 0.f;
  for (int i = threadIdx.x; i < kD; i += 256) {
    const float v = xr[i];
    s += v; ss += v * v;
  }
  __shared__ float r1[256], r2[256];
  r1[threadIdx.x] = s; r2[threadIdx.x] = ss;
  __syncthreads();
  for (int st = 128; st > 0; st >>= 1) {
    if (threadIdx.x < st) {
      r1[threadIdx.x] += r1[threadIdx.x + st];
      r2[threadIdx.x] += r2[threadIdx.x + st];
    }
    __syncthreads();
  }
  const float mu   = r1[0] * (1.f / kD);
  const float var  = r2[0] * (1.f / kD) - mu * mu;
  const float rstd = rsqrtf(var + 1e-5f);
  for (int i = threadIdx.x; i < kD; i += 256)
    out[(size_t)row * kD + i] = (__bf16)((xr[i] - mu) * rstd * w[i] + bvec[i]);
}

// ---------------------------------------------------------------------------
// WMMA GEMM: C = A[M,K]bf16 * W[K,N]bf16 + bias
//   block tile 128x64, BK=64; 8 waves each own a 32x32 tile (2x2 WMMA accs).
//   A tile staged via async-to-LDS DMA, double buffered; W tile staged
//   transposed ([n][k]) through VGPRs. Pipeline:
//     issue(t+1) -> WMMA(t) -> s_wait_asynccnt 0 -> barrier
// epilogue modes:
//   0: bf16 plain [M,N]
//   1: bf16 per-head scatter, ldo=128 (+off 0/64)   -> Q2/K2 [B,H,T,128]
//   2: bf16 per-head scatter, ldo=64                -> V     [B,H,T,64]
//   3: f32 [M,N] = v + resid
// ---------------------------------------------------------------------------
__global__ __launch_bounds__(256) void gemm_bf16_wmma_kernel(
    const __bf16* __restrict__ A, const __bf16* __restrict__ W,
    const float* __restrict__ bias, void* __restrict__ out,
    const float* __restrict__ resid, int M, int N, int K,
    int mode, int ldo, int off, int gelu) {
  constexpr int LDT = 72;                 // padded tile stride (elements)
  __shared__ __attribute__((aligned(16))) __bf16 As[2][128 * LDT];
  __shared__ __attribute__((aligned(16))) __bf16 Bs[2][64 * LDT];

  const int tid = threadIdx.x;
  const int colBase = blockIdx.x * 64;
  const int rowBase = blockIdx.y * 128;
  const int wv = tid >> 5;
  const int wm = (wv & 3) * 32;
  const int wn = (wv >> 2) * 32;

  v8f acc[2][2] = {{zero8(), zero8()}, {zero8(), zero8()}};

  const int am = tid >> 1, ak = (tid & 1) * 32;          // A: 64 B/thread
  const int wk = tid >> 2, wnoff = (tid & 3) * 16;       // W: 32 B/thread

  const unsigned ldsA[2] = {lds_addr32(&As[0][am * LDT + ak]),
                            lds_addr32(&As[1][am * LDT + ak])};

  auto stageA = [&](int k0, int buf) {   // DMA straight into LDS (ASYNCcnt)
    async_ld_64B(ldsA[buf], A + (size_t)(rowBase + am) * K + k0 + ak);
  };
  auto stageB = [&](int k0, int buf) {   // transpose through VGPRs
    const uint4* g = (const uint4*)(W + (size_t)(k0 + wk) * N + colBase + wnoff);
    __bf16 tmp[16];
    *(uint4*)tmp       = g[0];
    *(uint4*)(tmp + 8) = g[1];
#pragma unroll
    for (int i = 0; i < 16; ++i) Bs[buf][(wnoff + i) * LDT + wk] = tmp[i];
    if (k0 + 64 < K)   // speculative prefetch -> global_prefetch_b8
      __builtin_prefetch(W + (size_t)(k0 + 64 + wk) * N + colBase + wnoff, 0, 3);
  };

  stageA(0, 0);
  stageB(0, 0);
  wait_async0();
  __syncthreads();

  const int NT = K >> 6;                 // K / 64 tiles
  for (int t = 0; t < NT; ++t) {
    const int buf = t & 1;
    if (t + 1 < NT) {                    // issue next tile while computing
      stageA((t + 1) * 64, buf ^ 1);
      stageB((t + 1) * 64, buf ^ 1);
    }
#pragma unroll
    for (int kk = 0; kk < 64; kk += 32) {
      const v16bf a0 = load_frag16(&As[buf][(wm + 0)  * LDT + kk], LDT);
      const v16bf a1 = load_frag16(&As[buf][(wm + 16) * LDT + kk], LDT);
      const v16bf b0 = load_frag16(&Bs[buf][(wn + 0)  * LDT + kk], LDT);
      const v16bf b1 = load_frag16(&Bs[buf][(wn + 16) * LDT + kk], LDT);
      acc[0][0] = wmma_bf16(a0, b0, acc[0][0]);
      acc[0][1] = wmma_bf16(a0, b1, acc[0][1]);
      acc[1][0] = wmma_bf16(a1, b0, acc[1][0]);
      acc[1][1] = wmma_bf16(a1, b1, acc[1][1]);
    }
    wait_async0();                       // next-tile DMA landed
    __syncthreads();                     // everyone done reading buf
  }

  // epilogue (C/D layout: VGPR r holds rows r / r+8; lane%16 = column)
  const int lane = tid & 31;
  const int cn = lane & 15, rh = (lane >> 4) * 8;
#pragma unroll
  for (int i = 0; i < 2; ++i)
#pragma unroll
    for (int j = 0; j < 2; ++j)
#pragma unroll
      for (int r = 0; r < 8; ++r) {
        const int row = rowBase + wm + 16 * i + r + rh;
        const int col = colBase + wn + 16 * j + cn;
        float v = acc[i][j][r] + bias[col];
        if (gelu) v = 0.5f * v * (1.f + erff(v * 0.70710678f));  // exact gelu
        if (mode == 3) {
          const size_t idx = (size_t)row * N + col;
          ((float*)out)[idx] = v + resid[idx];
        } else if (mode == 0) {
          ((__bf16*)out)[(size_t)row * N + col] = (__bf16)v;
        } else {
          const int bb = row >> 11, t2 = row & (kT - 1);
          const int hh = col >> 6,  dh = col & (kDH - 1);
          const size_t dst =
              (((size_t)(bb * kH + hh) * kT + t2) * (size_t)ldo) + dh + off;
          ((__bf16*)out)[dst] = (__bf16)v;
        }
      }
}

// ---------------------------------------------------------------------------
// Flash attention with combined head-dim 128 (content||structure):
//   per block: one (b,h) and a 64-query tile; loop over 32-key tiles with
//   online softmax. S and P*V both via WMMA. Q/K tiles staged via async DMA.
//   Causal mask applied analytically (mask input unused).
// ---------------------------------------------------------------------------
__global__ __launch_bounds__(256) void attn_flash_wmma_kernel(
    const __bf16* __restrict__ Q2, const __bf16* __restrict__ K2,
    const __bf16* __restrict__ Vh, __bf16* __restrict__ aO) {
  constexpr int LQ = 136;   // 128 + pad
  constexpr int LS = 40;    // 32 + pad
  __shared__ __attribute__((aligned(16))) __bf16 Qs[64 * LQ];  // 64 x 128
  __shared__ __attribute__((aligned(16))) __bf16 Ks[32 * LQ];  // 32 x 128
  __shared__ float  Sf[64 * LS];   // scores  64 x 32
  __shared__ __bf16 Ps[64 * LS];   // probs   64 x 32
  __shared__ __bf16 Vt[64 * LS];   // V^T     64(dh) x 32(key) ([n][k])
  __shared__ float  mS[64], lS[64], sS[64];

  const int tid  = threadIdx.x;
  const int lane = tid & 31;
  const int wv   = tid >> 5;
  const int qb   = blockIdx.x;          // query tile
  const int bh   = blockIdx.y;          // b*H + h
  const int b    = bh >> 4;
  const int h    = bh & 15;

  // async-stage the 64x128 query tile once (64 B per thread)
  {
    const int row = tid >> 2, c0 = (tid & 3) * 32;
    async_ld_64B(lds_addr32(&Qs[row * LQ + c0]),
                 Q2 + (((size_t)bh * kT) + qb * 64 + row) * 128 + c0);
  }
  if (tid < 64) { mS[tid] = -1e30f; lS[tid] = 0.f; }

  const int om = (wv & 3) * 16;         // O tile: wave = 16 rows x 32 dh
  const int on = (wv >> 2) * 32;
  v8f o0 = zero8(), o1 = zero8();

  const int sm = (wv & 3) * 16;         // S tile: wave = 16 q x 16 k
  const int sn = (wv >> 2) * 16;
  const int cn = lane & 15, rh = (lane >> 4) * 8;

  const int kr = tid >> 3, kc0 = (tid & 7) * 16;   // K tile: 32 B per thread
  const unsigned ldsK = lds_addr32(&Ks[kr * LQ + kc0]);

  wait_async0();
  __syncthreads();

  const int ktMax = 2 * qb + 1;         // causal: keys <= query-tile end
  for (int kt = 0; kt <= ktMax; ++kt) {
    // async-stage K tile (natural layout == [n][k] for WMMA B of Q*K^T)
    async_ld_32B(ldsK, K2 + (((size_t)bh * kT) + kt * 32 + kr) * 128 + kc0);
    // stage V tile transposed: Vt[dh][key]
    {
      const int key = tid >> 3, d0 = (tid & 7) * 8;
      const uint4* g =
          (const uint4*)(Vh + (((size_t)bh * kT) + kt * 32 + key) * 64 + d0);
      __bf16 tmp[8];
      *(uint4*)tmp = g[0];
#pragma unroll
      for (int i = 0; i < 8; ++i) Vt[(d0 + i) * LS + key] = tmp[i];
    }
    wait_async0();
    __syncthreads();

    // S = (Q * K^T) / sqrt(64) over combined 128-dim head
    v8f sacc = zero8();
#pragma unroll
    for (int kk = 0; kk < 128; kk += 32) {
      const v16bf a  = load_frag16(Qs + sm * LQ + kk, LQ);
      const v16bf bf = load_frag16(Ks + sn * LQ + kk, LQ);
      sacc = wmma_bf16(a, bf, sacc);
    }
#pragma unroll
    for (int r = 0; r < 8; ++r)
      Sf[(sm + r + rh) * LS + sn + cn] = sacc[r] * 0.125f;
    __syncthreads();

    // online softmax: 4 threads per row, 8 cols each
    {
      const int row = tid >> 2, sub = tid & 3;
      const int qglob = qb * 64 + row;
      float vloc[8];
      float mx = -1e30f;
#pragma unroll
      for (int c = 0; c < 8; ++c) {
        const int col = sub * 8 + c;
        float v = Sf[row * LS + col];
        if (kt * 32 + col > qglob) v = -1e30f;   // causal mask
        vloc[c] = v;
        mx = fmaxf(mx, v);
      }
      mx = fmaxf(mx, __shfl_xor(mx, 1, 32));
      mx = fmaxf(mx, __shfl_xor(mx, 2, 32));
      const float mold = mS[row];
      const float mnew = fmaxf(mold, mx);
      float sum = 0.f;
#pragma unroll
      for (int c = 0; c < 8; ++c) {
        const float pe = __expf(vloc[c] - mnew);
        Ps[row * LS + sub * 8 + c] = (__bf16)pe;
        sum += pe;
      }
      sum += __shfl_xor(sum, 1, 32);
      sum += __shfl_xor(sum, 2, 32);
      if (sub == 0) {
        const float sc = __expf(mold - mnew);
        mS[row] = mnew;
        lS[row] = lS[row] * sc + sum;
        sS[row] = sc;
      }
    }
    __syncthreads();

    // O = diag(exp(mold-mnew)) * O + P * V
#pragma unroll
    for (int r = 0; r < 8; ++r) {
      const float sc = sS[om + r + rh];
      o0[r] *= sc; o1[r] *= sc;
    }
    {
      const v16bf pa = load_frag16(Ps + om * LS, LS);
      const v16bf v0 = load_frag16(Vt + on * LS, LS);
      const v16bf v1 = load_frag16(Vt + (on + 16) * LS, LS);
      o0 = wmma_bf16(pa, v0, o0);
      o1 = wmma_bf16(pa, v1, o1);
    }
    __syncthreads();
  }

  // normalize and scatter back to [B*T, D] bf16
#pragma unroll
  for (int r = 0; r < 8; ++r) {
    const int row = om + r + rh;
    const float inv = 1.f / lS[row];
    const size_t grow = (size_t)b * kT + qb * 64 + row;
    aO[grow * kD + h * kDH + (on + cn)]      = (__bf16)(o0[r] * inv);
    aO[grow * kD + h * kDH + (on + 16 + cn)] = (__bf16)(o1[r] * inv);
  }
}

// ---------------------------------------------------------------------------
// launcher
// ---------------------------------------------------------------------------
extern "C" void kernel_launch(void* const* d_in, const int* in_sizes, int n_in,
                              void* d_out, int out_size, void* d_ws,
                              size_t ws_size, hipStream_t stream) {
  (void)in_sizes; (void)n_in; (void)out_size; (void)ws_size;
  const float* x    = (const float*)d_in[0];
  const float* pos  = (const float*)d_in[1];
  const float* Wqc  = (const float*)d_in[2];
  const float* bqc  = (const float*)d_in[3];
  const float* Wkc  = (const float*)d_in[4];
  const float* bkc  = (const float*)d_in[5];
  const float* Wqs  = (const float*)d_in[6];
  const float* bqs  = (const float*)d_in[7];
  const float* Wks  = (const float*)d_in[8];
  const float* bks  = (const float*)d_in[9];
  const float* Wv   = (const float*)d_in[10];
  const float* bv   = (const float*)d_in[11];
  const float* Wo   = (const float*)d_in[12];
  const float* bo   = (const float*)d_in[13];
  const float* ln1w = (const float*)d_in[14];
  const float* ln1b = (const float*)d_in[15];
  const float* ln2w = (const float*)d_in[16];
  const float* ln2b = (const float*)d_in[17];
  const float* Wf1  = (const float*)d_in[18];
  const float* bf1  = (const float*)d_in[19];
  const float* Wf2  = (const float*)d_in[20];
  const float* bf2  = (const float*)d_in[21];
  // d_in[22] causal_mask: applied analytically in the attention kernel.

  char* p = (char*)d_ws;
  auto alloc = [&](size_t n) -> char* {
    char* r = p;
    p += (n + 255) & ~(size_t)255;
    return r;
  };
  __bf16* hb   = (__bf16*)alloc((size_t)kBT * kD * 2);        // LN1(x)
  __bf16* pb   = (__bf16*)alloc((size_t)kBT * kD * 2);        // pos_emb
  __bf16* wqcB = (__bf16*)alloc((size_t)kD * kD * 2);
  __bf16* wkcB = (__bf16*)alloc((size_t)kD * kD * 2);
  __bf16* wqsB = (__bf16*)alloc((size_t)kD * kD * 2);
  __bf16* wksB = (__bf16*)alloc((size_t)kD * kD * 2);
  __bf16* wvB  = (__bf16*)alloc((size_t)kD * kD * 2);
  __bf16* woB  = (__bf16*)alloc((size_t)kD * kD * 2);
  __bf16* wf1B = (__bf16*)alloc((size_t)kD * kF * 2);
  __bf16* wf2B = (__bf16*)alloc((size_t)kF * kD * 2);
  __bf16* Q2   = (__bf16*)alloc((size_t)kB * kH * kT * 128 * 2);  // [Qc|Qs]
  __bf16* K2   = (__bf16*)alloc((size_t)kB * kH * kT * 128 * 2);  // [Kc|Ks]
  __bf16* Vh   = (__bf16*)alloc((size_t)kB * kH * kT * kDH * 2);
  __bf16* aO   = (__bf16*)alloc((size_t)kBT * kD * 2);
  float*  x1   = (float*)alloc((size_t)kBT * kD * 4);          // residual 1
  __bf16* h2   = (__bf16*)alloc((size_t)kBT * kD * 2);         // LN2(x1)
  __bf16* f1   = (__bf16*)alloc((size_t)kBT * kF * 2);         // gelu(h2@Wf1)

  const dim3 blk(256);
  const dim3 gCast(1024);

  // ---- precision staging: fp32 -> bf16 ----
  cast_f32_bf16_kernel<<<gCast, blk, 0, stream>>>(pos, pb, kBT * kD);
  cast_f32_bf16_kernel<<<gCast, blk, 0, stream>>>(Wqc, wqcB, kD * kD);
  cast_f32_bf16_kernel<<<gCast, blk, 0, stream>>>(Wkc, wkcB, kD * kD);
  cast_f32_bf16_kernel<<<gCast, blk, 0, stream>>>(Wqs, wqsB, kD * kD);
  cast_f32_bf16_kernel<<<gCast, blk, 0, stream>>>(Wks, wksB, kD * kD);
  cast_f32_bf16_kernel<<<gCast, blk, 0, stream>>>(Wv,  wvB,  kD * kD);
  cast_f32_bf16_kernel<<<gCast, blk, 0, stream>>>(Wo,  woB,  kD * kD);
  cast_f32_bf16_kernel<<<gCast, blk, 0, stream>>>(Wf1, wf1B, kD * kF);
  cast_f32_bf16_kernel<<<gCast, blk, 0, stream>>>(Wf2, wf2B, kF * kD);

  // ---- LN1 ----
  ln_bf16_kernel<<<dim3(kBT), blk, 0, stream>>>(x, ln1w, ln1b, hb);

  // ---- projections (scatter straight into per-head layouts) ----
  const dim3 gP(kD / 64, kBT / 128);
  gemm_bf16_wmma_kernel<<<gP, blk, 0, stream>>>(hb, wqcB, bqc, Q2, nullptr,
                                                kBT, kD, kD, 1, 128, 0,  0);
  gemm_bf16_wmma_kernel<<<gP, blk, 0, stream>>>(hb, wkcB, bkc, K2, nullptr,
                                                kBT, kD, kD, 1, 128, 0,  0);
  gemm_bf16_wmma_kernel<<<gP, blk, 0, stream>>>(pb, wqsB, bqs, Q2, nullptr,
                                                kBT, kD, kD, 1, 128, 64, 0);
  gemm_bf16_wmma_kernel<<<gP, blk, 0, stream>>>(pb, wksB, bks, K2, nullptr,
                                                kBT, kD, kD, 1, 128, 64, 0);
  gemm_bf16_wmma_kernel<<<gP, blk, 0, stream>>>(hb, wvB, bv, Vh, nullptr,
                                                kBT, kD, kD, 2, 64, 0,  0);

  // ---- flash attention over combined 128-dim head ----
  attn_flash_wmma_kernel<<<dim3(kT / 64, kB * kH), blk, 0, stream>>>(Q2, K2,
                                                                     Vh, aO);

  // ---- output projection + residual 1 (f32) ----
  gemm_bf16_wmma_kernel<<<gP, blk, 0, stream>>>(aO, woB, bo, x1, x,
                                                kBT, kD, kD, 3, 0, 0, 0);

  // ---- LN2 + FFN ----
  ln_bf16_kernel<<<dim3(kBT), blk, 0, stream>>>(x1, ln2w, ln2b, h2);
  const dim3 gF(kF / 64, kBT / 128);
  gemm_bf16_wmma_kernel<<<gF, blk, 0, stream>>>(h2, wf1B, bf1, f1, nullptr,
                                                kBT, kF, kD, 0, 0, 0, 1);
  gemm_bf16_wmma_kernel<<<gP, blk, 0, stream>>>(f1, wf2B, bf2, d_out, x1,
                                                kBT, kD, kF, 3, 0, 0, 0);
}
